// VIT_52776558133441
// MI455X (gfx1250) — compile-verified, matrix operand order checked by reading, not run on hardware
//
#include <hip/hip_runtime.h>
#include <hip/hip_bf16.h>

typedef __attribute__((ext_vector_type(16))) __bf16 v16bf;
typedef __attribute__((ext_vector_type(8)))  __bf16 v8bf;
typedef __attribute__((ext_vector_type(8)))  float  v8f;
typedef __attribute__((ext_vector_type(4)))  unsigned int v4u;
typedef __attribute__((ext_vector_type(8)))  int v8i;
typedef __attribute__((ext_vector_type(4)))  int v4i;

#define N_LAYERS   5
#define DMODEL     768
#define SEQ        17
#define IMG_PER_BLK 2
#define ROWS       (SEQ * IMG_PER_BLK)   // 34 tokens per block
#define MPAD       48                    // 3 row-tiles of 16
#define LDXN       776                   // 768 + 8 pad (bank-conflict dodge)
#define LDQKV      1032                  // 1024 + 8 pad (q:0..767 k:768..895 v:896..1023)
#define LDM        264                   // 256 + 8 pad
#define HEADS_Q    12
#define GQA_GROUP  6

#if __has_builtin(__builtin_amdgcn_tensor_load_to_lds) && __has_builtin(__builtin_amdgcn_tensor_store_from_lds)
#define HAVE_TDM 1
#else
#define HAVE_TDM 0
#endif

// ---------------------------------------------------------------------------
// WMMA helpers (CDNA5 16x16x32 bf16, fp32 accumulate)
// ---------------------------------------------------------------------------
__device__ inline v8f wmma_bf16(v16bf a, v16bf b, v8f c) {
    return __builtin_amdgcn_wmma_f32_16x16x32_bf16(
        /*neg_a=*/false, a, /*neg_b=*/false, b,
        /*c_mod=*/(short)0, c, /*reuse_a=*/false, /*reuse_b=*/false);
}

// A fragment: 16x32 tile from LDS, row-major with stride lda.
// lane%16 = row; lanes 0-15 hold K {kb..kb+7, kb+16..kb+23},
// lanes 16-31 hold {kb+8..kb+15, kb+24..kb+31}.
__device__ inline v16bf load_frag_a(const __bf16* base, int lda, int mrow0, int kb, int lane) {
    int m  = mrow0 + (lane & 15);
    int kh = lane >> 4;
    const __bf16* p = base + m * lda + kb + kh * 8;
    v8bf lo = *(const v8bf*)(p);
    v8bf hi = *(const v8bf*)(p + 16);
    return __builtin_shufflevector(lo, hi, 0,1,2,3,4,5,6,7,8,9,10,11,12,13,14,15);
}

// B fragment: pre-swizzled weights; each lane's 16 bf16 are contiguous (32B load).
__device__ inline v16bf load_frag_b(const __bf16* bsw, int ktiles, int kt, int nt, int lane) {
    const __bf16* p = bsw + (((size_t)(nt * ktiles + kt)) * 32 + lane) * 16;
    return *(const v16bf*)p;
}

// ---------------------------------------------------------------------------
// Weight swizzle: fp32 [K][N] row-major -> bf16 WMMA-fragment tiles
// ---------------------------------------------------------------------------
__global__ __launch_bounds__(256)
void vit_swizzle_kernel(const float* __restrict__ W, __bf16* __restrict__ out, int K, int N) {
    int ktiles = K >> 5;
    long total = (long)ktiles * (N >> 4) * 32;   // lane-chunks
    for (long c = (long)blockIdx.x * 256 + threadIdx.x; c < total; c += (long)gridDim.x * 256) {
        int  lane = (int)(c & 31);
        long t    = c >> 5;
        int  kt   = (int)(t % ktiles);
        int  nt   = (int)(t / ktiles);
        int  n    = nt * 16 + (lane & 15);
        int  kh   = lane >> 4;
        __bf16* dst = out + c * 16;
        #pragma unroll
        for (int j = 0; j < 16; ++j) {
            int k = kt * 32 + kh * 8 + ((j < 8) ? j : (8 + j));  // hi half at +16
            dst[j] = (__bf16)W[(size_t)k * N + n];
        }
    }
}

// ---------------------------------------------------------------------------
// Patch embedding: x(4096,1,28,28) -> H(4096,17,768) bf16 (cls at token 0)
// ---------------------------------------------------------------------------
__global__ __launch_bounds__(256)
void vit_patch_kernel(const float* __restrict__ x, const float* __restrict__ Wp,
                      const float* __restrict__ cls, __bf16* __restrict__ H) {
    __shared__ float patch[16][52];
    int b = blockIdx.x, tid = threadIdx.x;
    for (int i = tid; i < 784; i += 256) {
        int row = i / 28, col = i % 28;
        patch[(row / 7) * 4 + (col / 7)][(row % 7) * 7 + (col % 7)] = x[(size_t)b * 784 + i];
    }
    __syncthreads();
    __bf16* out = H + (size_t)b * SEQ * DMODEL;
    for (int i = tid; i < DMODEL; i += 256) out[i] = (__bf16)cls[i];
    for (int e = tid; e < 16 * DMODEL; e += 256) {
        int pp = e / DMODEL, n = e % DMODEL;
        float s = 0.f;
        #pragma unroll 7
        for (int k = 0; k < 49; ++k) s += patch[pp][k] * Wp[k * DMODEL + n];
        out[(1 + pp) * DMODEL + n] = (__bf16)s;
    }
}

// ---------------------------------------------------------------------------
// LayerNorm rows<ROWS from bf16 h (stride DMODEL) into bf16 dst (stride LDXN)
// ---------------------------------------------------------------------------
__device__ void layernorm_rows(const __bf16* src, __bf16* dst,
                               const float* g, const float* b, int tid) {
    int wave = tid >> 5, lane = tid & 31;
    for (int r = wave; r < ROWS; r += 8) {
        const __bf16* row = src + r * DMODEL;
        float s = 0.f, ss = 0.f;
        for (int c = lane; c < DMODEL; c += 32) { float v = (float)row[c]; s += v; ss += v * v; }
        #pragma unroll
        for (int off = 16; off; off >>= 1) {
            s  += __shfl_xor(s,  off, 32);
            ss += __shfl_xor(ss, off, 32);
        }
        float mean = s * (1.f / DMODEL);
        float var  = ss * (1.f / DMODEL) - mean * mean;
        float rstd = rsqrtf(var + 1e-5f);
        __bf16* drow = dst + r * LDXN;
        for (int c = lane; c < DMODEL; c += 32) {
            float v = (float)row[c];
            drow[c] = (__bf16)(((v - mean) * rstd) * g[c] + b[c]);
        }
    }
}

// ---------------------------------------------------------------------------
// One transformer layer. Block = 2 images (34 tokens, padded to 48 rows).
// Row-tile (mt) fusion: one B fragment feeds 3 WMMAs -> 3x less global traffic.
// ---------------------------------------------------------------------------
__global__ __launch_bounds__(256)
void vit_layer_kernel(__bf16* __restrict__ H,
                      const __bf16* __restrict__ Wq, const __bf16* __restrict__ Wk,
                      const __bf16* __restrict__ Wv, const __bf16* __restrict__ Wo,
                      const __bf16* __restrict__ W1, const __bf16* __restrict__ W2,
                      const __bf16* __restrict__ W3,
                      const float* __restrict__ bq, const float* __restrict__ bk,
                      const float* __restrict__ bv, const float* __restrict__ bo,
                      const float* __restrict__ b1, const float* __restrict__ b2,
                      const float* __restrict__ b3,
                      const float* __restrict__ lg, const float* __restrict__ lb) {
    extern __shared__ char smem[];
    __bf16* hbuf  = (__bf16*)smem;                                        // 48x768  (34 used)
    __bf16* xnbuf = (__bf16*)(smem + (size_t)MPAD * DMODEL * 2);          // 48xLDXN (xn, then o)
    __bf16* qkv   = (__bf16*)(smem + (size_t)MPAD * (DMODEL + LDXN) * 2); // 48xLDQKV
    __bf16* mbuf  = qkv;                                                  // reuse q region, stride LDM

    int tid = threadIdx.x, lane = tid & 31, wave = tid >> 5;
    size_t gbase = (size_t)blockIdx.x * ROWS * DMODEL;

#if HAVE_TDM
    // --- Tensor Data Mover descriptor: 1-D tile, 6528 x 8B = 34x768 bf16 ---
    const unsigned NE = (unsigned)(ROWS * DMODEL * 2 / 8);
    unsigned long long ga = (unsigned long long)(H + gbase);
    v4u g0 = { 1u,                                     // count=1, user mode
               (unsigned)(unsigned long long)hbuf,     // lds_addr (bytes)
               (unsigned)(ga & 0xffffffffu),
               (unsigned)((ga >> 32) & 0x01ffffffu) | (2u << 30) };  // addr[56:32] | type=2
    v8i g1 = {};
    g1[0] = (3 << 16);                                 // data_size = 8B, wg mask = 0
    g1[1] = (int)((NE & 0xffffu) << 16);               // tensor_dim0[15:0]
    g1[2] = (int)(((NE >> 16) & 0xffffu) | (1u << 16)); // tensor_dim0[31:16] | tensor_dim1=1
    g1[3] = (int)(NE << 16);                           // tile_dim0
    g1[5] = (int)NE;                                   // tensor_dim0_stride
    v4i zz4 = {0, 0, 0, 0};
    v8i zz8 = {};
    if (wave == 0) {
        __builtin_amdgcn_tensor_load_to_lds(g0, g1, zz4, zz4, zz8, 0);
        __builtin_amdgcn_s_wait_tensorcnt(0);
    }
#else
    {
        const uint4* src = (const uint4*)(H + gbase);
        uint4* dst = (uint4*)hbuf;
        for (int i = tid; i < ROWS * DMODEL / 8; i += 256) dst[i] = src[i];
    }
#endif
    // zero xn pad rows (stay zero for the whole kernel)
    for (int i = tid; i < (MPAD - ROWS) * LDXN; i += 256) xnbuf[ROWS * LDXN + i] = (__bf16)0.f;
    __syncthreads();

    // ---- LN #1 ----
    layernorm_rows(hbuf, xnbuf, lg, lb, tid);
    __syncthreads();

    // ---- merged Q/K/V GEMM: 64 column-tile groups (48 Q, 8 K, 8 V), 8/wave ----
    for (int grp = wave; grp < 64; grp += 8) {
        const __bf16* B; const float* bias; int nt, cbase;
        if (grp < 48)      { B = Wq; bias = bq; nt = grp;      cbase = 0;   }
        else if (grp < 56) { B = Wk; bias = bk; nt = grp - 48; cbase = 768; }
        else               { B = Wv; bias = bv; nt = grp - 56; cbase = 896; }
        v8f a0 = {}, a1 = {}, a2 = {};
        for (int kt = 0; kt < 24; ++kt) {
            v16bf bf = load_frag_b(B, 24, kt, nt, lane);
            a0 = wmma_bf16(load_frag_a(xnbuf, LDXN,  0, kt * 32, lane), bf, a0);
            a1 = wmma_bf16(load_frag_a(xnbuf, LDXN, 16, kt * 32, lane), bf, a1);
            a2 = wmma_bf16(load_frag_a(xnbuf, LDXN, 32, kt * 32, lane), bf, a2);
        }
        int nl = nt * 16 + (lane & 15);
        int col = cbase + nl;
        int mr = (lane >> 4) * 8;
        float bi = bias[nl];
        #pragma unroll
        for (int r = 0; r < 8; ++r) {
            qkv[(mr + r) * LDQKV + col]      = (__bf16)(a0[r] + bi);
            qkv[(16 + mr + r) * LDQKV + col] = (__bf16)(a1[r] + bi);
            qkv[(32 + mr + r) * LDQKV + col] = (__bf16)(a2[r] + bi);
        }
    }
    __syncthreads();

    // ---- RoPE in place on q (12 heads) and k (2 heads) ----
    const float kLnThetaOver32 = 0.3381180713960389f;  // ln(50000)/32
    for (int task = tid; task < ROWS * 14 * 32; task += 256) {
        int p    = task & 31;
        int slot = (task >> 5) % 14;
        int r    = task / (14 * 32);
        int t    = r % SEQ;
        int col  = (slot < 12) ? (slot * 64 + 2 * p) : (768 + (slot - 12) * 64 + 2 * p);
        float ang = (float)t * __expf(-(float)p * kLnThetaOver32);
        float sn, cs; __sincosf(ang, &sn, &cs);
        __bf16* ptr = qkv + r * LDQKV + col;
        float re = (float)ptr[0], im = (float)ptr[1];
        ptr[0] = (__bf16)(re * cs - im * sn);
        ptr[1] = (__bf16)(re * sn + im * cs);
    }
    __syncthreads();

    // ---- GQA attention (seq=17): o written into xnbuf (xn is dead) ----
    for (int task = tid; task < IMG_PER_BLK * HEADS_Q * SEQ; task += 256) {
        int img = task / (HEADS_Q * SEQ);
        int rem = task % (HEADS_Q * SEQ);
        int qh  = rem / SEQ;
        int t   = rem % SEQ;
        int qrow = img * SEQ + t;
        int kv   = qh / GQA_GROUP;
        const __bf16* qp = qkv + qrow * LDQKV + qh * 64;
        float qreg[64];
        #pragma unroll
        for (int d = 0; d < 64; ++d) qreg[d] = (float)qp[d];
        float sc[SEQ]; float mx = -1e30f;
        #pragma unroll
        for (int u = 0; u < SEQ; ++u) {
            const __bf16* kp = qkv + (img * SEQ + u) * LDQKV + 768 + kv * 64;
            float s = 0.f;
            #pragma unroll
            for (int d = 0; d < 64; ++d) s += qreg[d] * (float)kp[d];
            sc[u] = s * 0.125f;                      // 1/sqrt(64)
            mx = fmaxf(mx, sc[u]);
        }
        float denom = 0.f;
        #pragma unroll
        for (int u = 0; u < SEQ; ++u) { sc[u] = __expf(sc[u] - mx); denom += sc[u]; }
        float inv = 1.f / denom;
        __bf16* op = xnbuf + qrow * LDXN + qh * 64;
        #pragma unroll 8
        for (int d = 0; d < 64; ++d) {
            float acc = 0.f;
            #pragma unroll
            for (int u = 0; u < SEQ; ++u)
                acc += sc[u] * (float)qkv[(img * SEQ + u) * LDQKV + 896 + kv * 64 + d];
            op[d] = (__bf16)(acc * inv);
        }
    }
    __syncthreads();

    // ---- Wo GEMM + residual: h += o @ Wo + bo ----
    for (int nt = wave; nt < 48; nt += 8) {
        v8f a0 = {}, a1 = {}, a2 = {};
        for (int kt = 0; kt < 24; ++kt) {
            v16bf bf = load_frag_b(Wo, 24, kt, nt, lane);
            a0 = wmma_bf16(load_frag_a(xnbuf, LDXN,  0, kt * 32, lane), bf, a0);
            a1 = wmma_bf16(load_frag_a(xnbuf, LDXN, 16, kt * 32, lane), bf, a1);
            a2 = wmma_bf16(load_frag_a(xnbuf, LDXN, 32, kt * 32, lane), bf, a2);
        }
        int n = nt * 16 + (lane & 15);
        int mr = (lane >> 4) * 8;
        float bi = bo[n];
        #pragma unroll
        for (int r = 0; r < 8; ++r) {
            int m0 = mr + r, m1 = 16 + mr + r, m2 = 32 + mr + r;
            float h0 = (float)hbuf[m0 * DMODEL + n];
            hbuf[m0 * DMODEL + n] = (__bf16)(h0 + a0[r] + bi);
            float h1 = (float)hbuf[m1 * DMODEL + n];
            hbuf[m1 * DMODEL + n] = (__bf16)(h1 + a1[r] + bi);
            if (m2 < ROWS) {
                float h2 = (float)hbuf[m2 * DMODEL + n];
                hbuf[m2 * DMODEL + n] = (__bf16)(h2 + a2[r] + bi);
            }
        }
    }
    __syncthreads();

    // ---- LN #2 into xnbuf ----
    layernorm_rows(hbuf, xnbuf, lg, lb, tid);
    __syncthreads();

    // ---- fused W1/W2 GEMMs + SiLU gate -> mbuf (A fragments shared) ----
    for (int nt = wave; nt < 16; nt += 8) {
        v8f p0 = {}, p1 = {}, p2 = {}, q0 = {}, q1 = {}, q2 = {};
        for (int kt = 0; kt < 24; ++kt) {
            v16bf af0 = load_frag_a(xnbuf, LDXN,  0, kt * 32, lane);
            v16bf af1 = load_frag_a(xnbuf, LDXN, 16, kt * 32, lane);
            v16bf af2 = load_frag_a(xnbuf, LDXN, 32, kt * 32, lane);
            v16bf bf1 = load_frag_b(W1, 24, kt, nt, lane);
            v16bf bf2 = load_frag_b(W2, 24, kt, nt, lane);
            p0 = wmma_bf16(af0, bf1, p0);  q0 = wmma_bf16(af0, bf2, q0);
            p1 = wmma_bf16(af1, bf1, p1);  q1 = wmma_bf16(af1, bf2, q1);
            p2 = wmma_bf16(af2, bf1, p2);  q2 = wmma_bf16(af2, bf2, q2);
        }
        int n = nt * 16 + (lane & 15);
        int mr = (lane >> 4) * 8;
        float c1 = b1[n], c2 = b2[n];
        #pragma unroll
        for (int r = 0; r < 8; ++r) {
            {
                int m = mr + r;
                float uv = q0[r] + c2;
                float sw = uv / (1.f + __expf(-uv));
                mbuf[m * LDM + n] = (m < ROWS) ? (__bf16)((p0[r] + c1) * sw) : (__bf16)0.f;
            }
            {
                int m = 16 + mr + r;
                float uv = q1[r] + c2;
                float sw = uv / (1.f + __expf(-uv));
                mbuf[m * LDM + n] = (m < ROWS) ? (__bf16)((p1[r] + c1) * sw) : (__bf16)0.f;
            }
            {
                int m = 32 + mr + r;
                float uv = q2[r] + c2;
                float sw = uv / (1.f + __expf(-uv));
                mbuf[m * LDM + n] = (m < ROWS) ? (__bf16)((p2[r] + c1) * sw) : (__bf16)0.f;
            }
        }
    }
    __syncthreads();

    // ---- W3 GEMM + residual: h += mbuf @ W3 + b3 ----
    for (int nt = wave; nt < 48; nt += 8) {
        v8f a0 = {}, a1 = {}, a2 = {};
        for (int kt = 0; kt < 8; ++kt) {
            v16bf bf = load_frag_b(W3, 8, kt, nt, lane);
            a0 = wmma_bf16(load_frag_a(mbuf, LDM,  0, kt * 32, lane), bf, a0);
            a1 = wmma_bf16(load_frag_a(mbuf, LDM, 16, kt * 32, lane), bf, a1);
            a2 = wmma_bf16(load_frag_a(mbuf, LDM, 32, kt * 32, lane), bf, a2);
        }
        int n = nt * 16 + (lane & 15);
        int mr = (lane >> 4) * 8;
        float bi = b3[n];
        #pragma unroll
        for (int r = 0; r < 8; ++r) {
            int m0 = mr + r, m1 = 16 + mr + r, m2 = 32 + mr + r;
            float h0 = (float)hbuf[m0 * DMODEL + n];
            hbuf[m0 * DMODEL + n] = (__bf16)(h0 + a0[r] + bi);
            float h1 = (float)hbuf[m1 * DMODEL + n];
            hbuf[m1 * DMODEL + n] = (__bf16)(h1 + a1[r] + bi);
            if (m2 < ROWS) {
                float h2 = (float)hbuf[m2 * DMODEL + n];
                hbuf[m2 * DMODEL + n] = (__bf16)(h2 + a2[r] + bi);
            }
        }
    }
    __syncthreads();

    // ---- store residual stream back ----
#if HAVE_TDM
    if (wave == 0) {
        __builtin_amdgcn_tensor_store_from_lds(g0, g1, zz4, zz4, zz8, 0);
        __builtin_amdgcn_s_wait_tensorcnt(0);
    }
#else
    {
        const uint4* src = (const uint4*)hbuf;
        uint4* dst = (uint4*)(H + gbase);
        for (int i = tid; i < ROWS * DMODEL / 8; i += 256) dst[i] = src[i];
    }
#endif
}

// ---------------------------------------------------------------------------
// Final LN of token 0 + 768x10 head
// ---------------------------------------------------------------------------
__global__ __launch_bounds__(256)
void vit_head_kernel(const __bf16* __restrict__ H, const float* __restrict__ g,
                     const float* __restrict__ bb, const float* __restrict__ Wout,
                     float* __restrict__ out) {
    __shared__ float row[DMODEL];
    __shared__ float red[18];
    __shared__ float acc10[10];
    int b = blockIdx.x, tid = threadIdx.x;
    const __bf16* src = H + (size_t)b * SEQ * DMODEL;   // token 0
    float s = 0.f, ss = 0.f;
    for (int c = tid; c < DMODEL; c += 256) {
        float v = (float)src[c]; row[c] = v; s += v; ss += v * v;
    }
    #pragma unroll
    for (int off = 16; off; off >>= 1) { s += __shfl_xor(s, off, 32); ss += __shfl_xor(ss, off, 32); }
    if ((tid & 31) == 0) { red[tid >> 5] = s; red[8 + (tid >> 5)] = ss; }
    for (int i = tid; i < 10; i += 256) acc10[i] = 0.f;
    __syncthreads();
    if (tid == 0) {
        float a = 0.f, q = 0.f;
        for (int w = 0; w < 8; ++w) { a += red[w]; q += red[8 + w]; }
        float mean = a * (1.f / DMODEL);
        red[16] = mean;
        red[17] = q * (1.f / DMODEL) - mean * mean;
    }
    __syncthreads();
    float mean = red[16], rstd = rsqrtf(red[17] + 1e-5f);
    float part[10];
    #pragma unroll
    for (int c = 0; c < 10; ++c) part[c] = 0.f;
    for (int d = tid; d < DMODEL; d += 256) {
        float xn = (row[d] - mean) * rstd * g[d] + bb[d];
        #pragma unroll
        for (int c = 0; c < 10; ++c) part[c] += xn * Wout[d * 10 + c];
    }
    #pragma unroll
    for (int c = 0; c < 10; ++c) {
        float v = part[c];
        #pragma unroll
        for (int off = 16; off; off >>= 1) v += __shfl_xor(v, off, 32);
        if ((tid & 31) == 0) atomicAdd(&acc10[c], v);
    }
    __syncthreads();
    if (tid < 10) out[(size_t)b * 10 + tid] = acc10[tid];
}

// ---------------------------------------------------------------------------
// Host launch
// ---------------------------------------------------------------------------
extern "C" void kernel_launch(void* const* d_in, const int* in_sizes, int n_in,
                              void* d_out, int out_size, void* d_ws, size_t ws_size,
                              hipStream_t stream) {
    const float* x     = (const float*)d_in[0];
    const float* Wp    = (const float*)d_in[1];
    const float* cls   = (const float*)d_in[2];
    const float* ln_g  = (const float*)d_in[3];
    const float* ln_b  = (const float*)d_in[4];
    const float* Wq    = (const float*)d_in[5];
    const float* bq    = (const float*)d_in[6];
    const float* Wk    = (const float*)d_in[7];
    const float* bk    = (const float*)d_in[8];
    const float* Wv    = (const float*)d_in[9];
    const float* bv    = (const float*)d_in[10];
    const float* Wo    = (const float*)d_in[11];
    const float* bo    = (const float*)d_in[12];
    const float* W1    = (const float*)d_in[13];
    const float* b1    = (const float*)d_in[14];
    const float* W2    = (const float*)d_in[15];
    const float* b2    = (const float*)d_in[16];
    const float* W3    = (const float*)d_in[17];
    const float* b3    = (const float*)d_in[18];
    const float* lnf_g = (const float*)d_in[19];
    const float* lnf_b = (const float*)d_in[20];
    const float* Wout  = (const float*)d_in[21];

    char* ws = (char*)d_ws;
    size_t off = 0;
    auto take = [&](size_t bytes) -> char* {
        char* p = ws + off;
        off += (bytes + 255) & ~(size_t)255;
        return p;
    };
    __bf16* Hbuf = (__bf16*)take((size_t)4096 * SEQ * DMODEL * 2);
    __bf16* sWq  = (__bf16*)take((size_t)N_LAYERS * 768 * 768 * 2);
    __bf16* sWk  = (__bf16*)take((size_t)N_LAYERS * 768 * 128 * 2);
    __bf16* sWv  = (__bf16*)take((size_t)N_LAYERS * 768 * 128 * 2);
    __bf16* sWo  = (__bf16*)take((size_t)N_LAYERS * 768 * 768 * 2);
    __bf16* sW1  = (__bf16*)take((size_t)N_LAYERS * 768 * 256 * 2);
    __bf16* sW2  = (__bf16*)take((size_t)N_LAYERS * 768 * 256 * 2);
    __bf16* sW3  = (__bf16*)take((size_t)N_LAYERS * 256 * 768 * 2);

    for (int l = 0; l < N_LAYERS; ++l) {
        vit_swizzle_kernel<<<128, 256, 0, stream>>>(Wq + (size_t)l * 768 * 768, sWq + (size_t)l * 768 * 768, 768, 768);
        vit_swizzle_kernel<<<64,  256, 0, stream>>>(Wk + (size_t)l * 768 * 128, sWk + (size_t)l * 768 * 128, 768, 128);
        vit_swizzle_kernel<<<64,  256, 0, stream>>>(Wv + (size_t)l * 768 * 128, sWv + (size_t)l * 768 * 128, 768, 128);
        vit_swizzle_kernel<<<128, 256, 0, stream>>>(Wo + (size_t)l * 768 * 768, sWo + (size_t)l * 768 * 768, 768, 768);
        vit_swizzle_kernel<<<64,  256, 0, stream>>>(W1 + (size_t)l * 768 * 256, sW1 + (size_t)l * 768 * 256, 768, 256);
        vit_swizzle_kernel<<<64,  256, 0, stream>>>(W2 + (size_t)l * 768 * 256, sW2 + (size_t)l * 768 * 256, 768, 256);
        vit_swizzle_kernel<<<64,  256, 0, stream>>>(W3 + (size_t)l * 256 * 768, sW3 + (size_t)l * 256 * 768, 256, 768);
    }

    vit_patch_kernel<<<4096, 256, 0, stream>>>(x, Wp, cls, Hbuf);

    size_t lds_bytes = (size_t)MPAD * (DMODEL + LDXN + LDQKV) * 2;  // 247,296 B
    for (int l = 0; l < N_LAYERS; ++l) {
        vit_layer_kernel<<<4096 / IMG_PER_BLK, 256, lds_bytes, stream>>>(
            Hbuf,
            sWq + (size_t)l * 768 * 768, sWk + (size_t)l * 768 * 128,
            sWv + (size_t)l * 768 * 128, sWo + (size_t)l * 768 * 768,
            sW1 + (size_t)l * 768 * 256, sW2 + (size_t)l * 768 * 256,
            sW3 + (size_t)l * 256 * 768,
            bq + (size_t)l * 768, bk + (size_t)l * 128, bv + (size_t)l * 128,
            bo + (size_t)l * 768, b1 + (size_t)l * 256, b2 + (size_t)l * 256,
            b3 + (size_t)l * 768,
            ln_g + (size_t)l * 768, ln_b + (size_t)l * 768);
    }

    vit_head_kernel<<<4096, 256, 0, stream>>>(Hbuf, lnf_g, lnf_b, Wout, (float*)d_out);
}
